// QuasarAttention_27118423507037
// MI455X (gfx1250) — matrix operand order repaired and checked by reading, not compiled, not run on hardware
//
#include <hip/hip_runtime.h>

typedef __attribute__((ext_vector_type(16))) __bf16 v16bf;
typedef __attribute__((ext_vector_type(8)))  float  v8f;

union FragU { unsigned int u[8]; v16bf v; };

// async global->LDS 16B copy (per lane), tracked by ASYNCcnt
__device__ __forceinline__ void async_copy_b128(const unsigned int* lds_ptr,
                                                const __bf16* gptr) {
  const unsigned lds_off = (unsigned)(unsigned long long)(uintptr_t)lds_ptr;
  asm volatile("global_load_async_to_lds_b128 %0, %1, off"
               :: "v"(lds_off), "v"((unsigned long long)(uintptr_t)gptr)
               : "memory");
}
__device__ __forceinline__ void wait_async_le4() {
  asm volatile("s_wait_asynccnt 0x4" ::: "memory");
}
__device__ __forceinline__ void wait_async_le0() {
  asm volatile("s_wait_asynccnt 0x0" ::: "memory");
}

// ---------------------------------------------------------------------------
// elementwise f32 -> bf16 cast
// ---------------------------------------------------------------------------
__global__ __launch_bounds__(256) void cast_f32_to_bf16(const float* __restrict__ in,
                                                        __bf16* __restrict__ out, int n) {
  int i = blockIdx.x * blockDim.x + threadIdx.x;
  if (i < n) out[i] = (__bf16)in[i];
}

// ---------------------------------------------------------------------------
// C[M,N] = A[M,K] * W[N,K]^T   (bf16 in, f32 accumulate via WMMA)
// block tile 128x128, K-step 32, 256 threads = 8 waves, each wave 4x2 16x16 tiles.
// Double-buffered LDS fed by global_load_async_to_lds_b128 (ASYNCcnt pipelined).
// LDS row stride = 20 dwords: 16B-aligned chunks + conflict-free fragment gathers.
// ---------------------------------------------------------------------------
#define LSTRIDE 20

template<bool OUT_BF16>
__global__ __launch_bounds__(256) void gemm_bf16_wmma(
    const __bf16* __restrict__ A,   // [M,K] row-major
    const __bf16* __restrict__ W,   // [N,K] row-major
    void* __restrict__ Cv, int M, int N, int K) {
  (void)M;
  __shared__ unsigned int As[2][128 * LSTRIDE];
  __shared__ unsigned int Ws[2][128 * LSTRIDE];

  const int tid  = threadIdx.x;
  const int lane = tid & 31;
  const int wave = tid >> 5;
  const int wm   = (wave & 1) * 64;   // wave M offset inside block tile
  const int wn   = (wave >> 1) * 32;  // wave N offset inside block tile
  const int hsel = lane >> 4;         // lane half (0/1)
  const int mr   = lane & 15;

  const int bm = blockIdx.y * 128;
  const int bn = blockIdx.x * 128;

  const int lrow = tid >> 2;          // 0..63
  const int lcol = (tid & 3) * 8;     // bf16 element col: 0,8,16,24
  const int lu   = lcol >> 1;         // dword col: 0,4,8,12 (16B aligned chunks)

  // per-thread global source rows (advance by k-step columns)
  const __bf16* Ag0 = A + (size_t)(bm + lrow)      * K + lcol;
  const __bf16* Ag1 = A + (size_t)(bm + lrow + 64) * K + lcol;
  const __bf16* Wg0 = W + (size_t)(bn + lrow)      * K + lcol;
  const __bf16* Wg1 = W + (size_t)(bn + lrow + 64) * K + lcol;

  v8f acc[4][2] = {};

  // A-fragment dword gather (16-bit A 16x32 layout):
  // k_pair = 2*(p&3) + (p>>2)*16 + hsel*8 -> dword index:
  int kA[8];
#pragma unroll
  for (int p = 0; p < 8; ++p) kA[p] = (p & 3) + (p >> 2) * 8 + hsel * 4;

  const int NT = K >> 5;  // number of 32-wide K tiles

  // prefetch tile 0 into buffer 0
  async_copy_b128(&As[0][(lrow)      * LSTRIDE + lu], Ag0);
  async_copy_b128(&As[0][(lrow + 64) * LSTRIDE + lu], Ag1);
  async_copy_b128(&Ws[0][(lrow)      * LSTRIDE + lu], Wg0);
  async_copy_b128(&Ws[0][(lrow + 64) * LSTRIDE + lu], Wg1);

  for (int it = 0; it < NT; ++it) {
    const int cur = it & 1;
    __syncthreads();  // all reads of the buffer we are about to overwrite are done
    if (it + 1 < NT) {
      const int nxt = cur ^ 1;
      const int ko = (it + 1) << 5;
      async_copy_b128(&As[nxt][(lrow)      * LSTRIDE + lu], Ag0 + ko);
      async_copy_b128(&As[nxt][(lrow + 64) * LSTRIDE + lu], Ag1 + ko);
      async_copy_b128(&Ws[nxt][(lrow)      * LSTRIDE + lu], Wg0 + ko);
      async_copy_b128(&Ws[nxt][(lrow + 64) * LSTRIDE + lu], Wg1 + ko);
      wait_async_le4();  // tile `it`'s 4 copies retired; only the 4 new in flight
    } else {
      wait_async_le0();
    }
    __syncthreads();  // tile `it` visible to whole workgroup

    const unsigned int* Ab = As[cur];
    const unsigned int* Wb = Ws[cur];
    FragU af[4], wf[2];
#pragma unroll
    for (int i = 0; i < 4; ++i) {
      const int row = wm + i * 16 + mr;   // lane holds A row m = mr
#pragma unroll
      for (int p = 0; p < 8; ++p) af[i].u[p] = Ab[row * LSTRIDE + kA[p]];
    }
#pragma unroll
    for (int j = 0; j < 2; ++j) {
      const int row = wn + j * 16 + mr;   // lane holds B col n = mr; same K order as A (kA)
#pragma unroll
      for (int p = 0; p < 8; ++p) wf[j].u[p] = Wb[row * LSTRIDE + kA[p]];
    }

#pragma unroll
    for (int i = 0; i < 4; ++i)
#pragma unroll
      for (int j = 0; j < 2; ++j)
        acc[i][j] = __builtin_amdgcn_wmma_f32_16x16x32_bf16(
            false, af[i].v, false, wf[j].v, (short)0, acc[i][j], false, false);
  }

  // C/D layout: VGPR r -> m = r + 8*half ; lane&15 -> n
#pragma unroll
  for (int i = 0; i < 4; ++i)
#pragma unroll
    for (int j = 0; j < 2; ++j)
#pragma unroll
      for (int r = 0; r < 8; ++r) {
        const int row = bm + wm + i * 16 + r + hsel * 8;
        const int col = bn + wn + j * 16 + mr;
        const float val = acc[i][j][r];
        if constexpr (OUT_BF16) ((__bf16*)Cv)[(size_t)row * N + col] = (__bf16)val;
        else                    ((float*)Cv)[(size_t)row * N + col]  = val;
      }
}

// ---------------------------------------------------------------------------
// causal depthwise conv (4 taps) + SiLU on fused qkv buffer; writes q/k/v in
// [B, H, T, Dh] layout for the recurrence
// ---------------------------------------------------------------------------
__global__ __launch_bounds__(256) void conv_silu_kernel(
    const __bf16* __restrict__ qkv,  // [B, T, 6144]
    const float* __restrict__ qw, const float* __restrict__ kw, const float* __restrict__ vw,
    float* __restrict__ qc, float* __restrict__ kc, float* __restrict__ vc) {
  const int idx = blockIdx.x * blockDim.x + threadIdx.x;  // [0, B*T*2048)
  const int c = idx & 2047;
  const int t = (idx >> 11) & 4095;
  const int b = idx >> 23;
  const int h = c >> 7, d = c & 127;
  const size_t obase = ((((size_t)b * 16 + h) * 4096) + t) * 128 + d;

  float aq = 0.f, ak = 0.f, av = 0.f;
#pragma unroll
  for (int i = 0; i < 4; ++i) {
    const int tt = t - 3 + i;
    if (tt >= 0) {
      const size_t rb = ((size_t)b * 4096 + tt) * 6144;
      aq += (float)qkv[rb + c]        * qw[c * 4 + i];
      ak += (float)qkv[rb + 2048 + c] * kw[c * 4 + i];
      av += (float)qkv[rb + 4096 + c] * vw[c * 4 + i];
    }
  }
  qc[obase] = aq / (1.f + expf(-aq));
  kc[obase] = ak / (1.f + expf(-ak));
  vc[obase] = av / (1.f + expf(-av));
}

// ---------------------------------------------------------------------------
// delta-rule recurrence: one workgroup per (b,h); S[128][128] fp32 in LDS.
// thread t owns column v = t&127, k-range half (t>>7). Per step:
//   u = S^T k ; S += k (beta (v - u))^T ; o = scale * S^T q
// ---------------------------------------------------------------------------
__global__ __launch_bounds__(256) void delta_rule_kernel(
    const float* __restrict__ qc, const float* __restrict__ kc, const float* __restrict__ vc,
    const float* __restrict__ beta_log, float* __restrict__ o, int T) {
  extern __shared__ float smem[];
  float* S   = smem;               // 128*128
  float* qs  = smem + 128 * 128;   // 128
  float* ks  = qs + 128;           // 128
  float* vs  = ks + 128;           // 128
  float* wv  = vs + 128;           // 128
  float* red = wv + 128;           // 256

  const int tid = threadIdx.x;
  const int bh  = blockIdx.x;
  const int h   = bh & 15;
  const float beta  = log1pf(expf(beta_log[h]));   // softplus
  const float scale = rsqrtf(128.f);

  const size_t base = (size_t)bh * T * 128;
  const float* qp = qc + base;
  const float* kp = kc + base;
  const float* vp = vc + base;
  float* op = o + base;

  for (int i = tid; i < 128 * 128; i += 256) S[i] = 0.f;
  __syncthreads();

  const int v  = tid & 127;
  const int kb = (tid >> 7) * 64;

  for (int t = 0; t < T; ++t) {
    if (tid < 128) {
      qs[tid] = qp[(size_t)t * 128 + tid];
      ks[tid] = kp[(size_t)t * 128 + tid];
      vs[tid] = vp[(size_t)t * 128 + tid];
    }
    __syncthreads();
    float pu = 0.f;
#pragma unroll 8
    for (int k = kb; k < kb + 64; ++k) pu += ks[k] * S[k * 128 + v];
    red[tid] = pu;
    __syncthreads();
    if (tid < 128) wv[tid] = beta * (vs[tid] - (red[tid] + red[tid + 128]));
    __syncthreads();
    const float w = wv[v];
    float po = 0.f;
#pragma unroll 8
    for (int k = kb; k < kb + 64; ++k) {
      const float s = S[k * 128 + v] + ks[k] * w;
      S[k * 128 + v] = s;
      po += qs[k] * s;
    }
    red[tid] = po;
    __syncthreads();
    if (tid < 128) op[(size_t)t * 128 + tid] = scale * (red[tid] + red[tid + 128]);
    __syncthreads();
  }
}

// ---------------------------------------------------------------------------
// gated RMSNorm: warp (32 lanes) per (b,t,h) group of 128; out bf16 [B,T,2048]
// ---------------------------------------------------------------------------
__global__ __launch_bounds__(256) void gated_rmsnorm_kernel(
    const float* __restrict__ oatt,  // [B,16,T,128]
    const float* __restrict__ g,     // [B,T,2048]
    const float* __restrict__ nw,    // [128]
    __bf16* __restrict__ out) {      // [B,T,2048]
  const int gid  = blockIdx.x * 8 + (threadIdx.x >> 5);
  const int lane = threadIdx.x & 31;
  const int b = gid >> 16;            // T*H = 65536 groups per batch
  const int t = (gid >> 4) & 4095;
  const int h = gid & 15;
  const float* op = oatt + ((((size_t)b * 16 + h) * 4096) + t) * 128;
  float x[4], ss = 0.f;
#pragma unroll
  for (int i = 0; i < 4; ++i) { x[i] = op[lane + 32 * i]; ss += x[i] * x[i]; }
#pragma unroll
  for (int m = 16; m >= 1; m >>= 1) ss += __shfl_xor(ss, m, 32);
  const float rinv = rsqrtf(ss * (1.f / 128.f) + 1e-5f);
  const size_t ob = ((size_t)b * 4096 + t) * 2048 + h * 128;
#pragma unroll
  for (int i = 0; i < 4; ++i) {
    const int d = lane + 32 * i;
    const float gv = g[ob + d];
    out[ob + d] = (__bf16)(x[i] * rinv * nw[d] * (1.f / (1.f + expf(-gv))));
  }
}

// ---------------------------------------------------------------------------
extern "C" void kernel_launch(void* const* d_in, const int* in_sizes, int n_in,
                              void* d_out, int out_size, void* d_ws, size_t ws_size,
                              hipStream_t stream) {
  (void)in_sizes; (void)n_in; (void)out_size; (void)ws_size;
  const float* hs   = (const float*)d_in[0];
  const float* qkvw = (const float*)d_in[1];
  const float* qcw  = (const float*)d_in[2];
  const float* kcw  = (const float*)d_in[3];
  const float* vcw  = (const float*)d_in[4];
  const float* blog = (const float*)d_in[5];
  const float* gw   = (const float*)d_in[6];
  const float* onw  = (const float*)d_in[7];
  const float* opw  = (const float*)d_in[8];

  constexpr int B = 2, T = 4096, HID = 2048, D = 2048, M = B * T;

  char* p = (char*)d_ws;
  auto take = [&](size_t bytes) { char* r = p; p += (bytes + 255) & ~(size_t)255; return (void*)r; };
  __bf16* hsb   = (__bf16*)take((size_t)M * HID * 2);        // 32 MB
  __bf16* wqkvb = (__bf16*)take((size_t)3 * D * HID * 2);    // 24 MB
  __bf16* wgb   = (__bf16*)take((size_t)D * HID * 2);        //  8 MB
  __bf16* wob   = (__bf16*)take((size_t)HID * D * 2);        //  8 MB
  __bf16* qkvb  = (__bf16*)take((size_t)M * 3 * D * 2);      // 96 MB
  float*  gbuf  = (float*)take((size_t)M * D * 4);           // 64 MB
  float*  qc    = (float*)take((size_t)M * D * 4);           // 64 MB
  float*  kc    = (float*)take((size_t)M * D * 4);           // 64 MB
  float*  vc    = (float*)take((size_t)M * D * 4);           // 64 MB
  float*  oatt  = (float*)qkvb;   // reuse: qkv dead after conv
  __bf16* onrm  = (__bf16*)hsb;   // reuse: bf16 hidden dead after GEMMs

  const int TB = 256;
  cast_f32_to_bf16<<<(M * HID + TB - 1) / TB, TB, 0, stream>>>(hs, hsb, M * HID);
  cast_f32_to_bf16<<<(3 * D * HID + TB - 1) / TB, TB, 0, stream>>>(qkvw, wqkvb, 3 * D * HID);
  cast_f32_to_bf16<<<(D * HID + TB - 1) / TB, TB, 0, stream>>>(gw, wgb, D * HID);
  cast_f32_to_bf16<<<(HID * D + TB - 1) / TB, TB, 0, stream>>>(opw, wob, HID * D);

  // fused QKV projection (bf16 out) and gate projection (f32 out)
  gemm_bf16_wmma<true ><<<dim3(3 * D / 128, M / 128), TB, 0, stream>>>(hsb, wqkvb, qkvb, M, 3 * D, HID);
  gemm_bf16_wmma<false><<<dim3(D / 128, M / 128), TB, 0, stream>>>(hsb, wgb, gbuf, M, D, HID);

  conv_silu_kernel<<<(M * D) / TB, TB, 0, stream>>>(qkvb, qcw, kcw, vcw, qc, kc, vc);

  delta_rule_kernel<<<B * 16, TB, (128 * 128 + 4 * 128 + 256) * sizeof(float), stream>>>(
      qc, kc, vc, blog, oatt, T);

  gated_rmsnorm_kernel<<<(M * 16) / 8, TB, 0, stream>>>(oatt, gbuf, onw, onrm);

  gemm_bf16_wmma<false><<<dim3(HID / 128, M / 128), TB, 0, stream>>>(onrm, wob, (float*)d_out, M, HID, D);
}
